// ProteinMPNN_21741124452570
// MI455X (gfx1250) — matrix operand order repaired and checked by reading, not compiled, not run on hardware
//
#include <hip/hip_runtime.h>
#include <hip/hip_bf16.h>
#include <cstdint>

typedef _Float16 half_t;
typedef __attribute__((ext_vector_type(16))) _Float16 v16h;
typedef __attribute__((ext_vector_type(8)))  _Float16 v8h;
typedef __attribute__((ext_vector_type(8)))  float    v8f;

#define L_    2048
#define KNN_  30
#define E_    (L_*KNN_)
#define HID_  192
#define EDGE_ 324
#define MSGK_ 736   // 2*HID + EDGE = 708, padded to mult of 32
#define DH_   48
#define DHP_  64

// ======================= geometry helpers =======================
struct V3 { float x,y,z; };
__device__ __forceinline__ V3 v3load(const float* p, int i){ return {p[3*i],p[3*i+1],p[3*i+2]}; }
__device__ __forceinline__ V3 vsub(V3 a, V3 b){ return {a.x-b.x,a.y-b.y,a.z-b.z}; }
__device__ __forceinline__ float vdot(V3 a, V3 b){ return a.x*b.x+a.y*b.y+a.z*b.z; }
__device__ __forceinline__ V3 vcross(V3 a, V3 b){ return {a.y*b.z-a.z*b.y, a.z*b.x-a.x*b.z, a.x*b.y-a.y*b.x}; }
__device__ __forceinline__ V3 vunit(V3 a){ float n = sqrtf(vdot(a,a)) + 1e-8f; return {a.x/n,a.y/n,a.z/n}; }
__device__ __forceinline__ float dihedral(V3 p0, V3 p1, V3 p2, V3 p3){
  V3 b1=vsub(p1,p0), b2=vsub(p2,p1), b3=vsub(p3,p2);
  V3 n1=vunit(vcross(b1,b2)), n2=vunit(vcross(b2,b3));
  V3 m1=vcross(n1, vunit(b2));
  return atan2f(vdot(m1,n2), vdot(n1,n2));
}

// ======================= kNN (top-30 by CA distance) =======================
__global__ void knn_kernel(const float* __restrict__ ca, int* __restrict__ dstIdx) {
  __shared__ float d2[L_];
  __shared__ float rv[256];
  __shared__ int   ri[256];
  int i = blockIdx.x, t = threadIdx.x;
  float xi = ca[3*i], yi = ca[3*i+1], zi = ca[3*i+2];
  for (int j = t; j < L_; j += 256) {
    float dx = xi - ca[3*j], dy = yi - ca[3*j+1], dz = zi - ca[3*j+2];
    d2[j] = (j == i) ? 3.4e38f : (dx*dx + dy*dy + dz*dz);
  }
  __syncthreads();
  for (int it = 0; it < KNN_; ++it) {
    float best = 3.39e38f; int bi = L_;
    for (int j = t; j < L_; j += 256) {
      float v = d2[j];
      if (v < best || (v == best && j < bi)) { best = v; bi = j; }
    }
    rv[t] = best; ri[t] = bi;
    __syncthreads();
    for (int s = 128; s > 0; s >>= 1) {
      if (t < s) {
        if (rv[t+s] < rv[t] || (rv[t+s] == rv[t] && ri[t+s] < ri[t])) { rv[t] = rv[t+s]; ri[t] = ri[t+s]; }
      }
      __syncthreads();
    }
    if (t == 0) { dstIdx[i*KNN_ + it] = ri[0]; d2[ri[0]] = 3.4e38f; }
    __syncthreads();
  }
}

// ======================= node features [L,15] =======================
__global__ void node_feats_kernel(const float* __restrict__ cN, const float* __restrict__ cCA,
                                  const float* __restrict__ cC, float* __restrict__ nf) {
  int i = blockIdx.x*blockDim.x + threadIdx.x;
  if (i >= L_) return;
  V3 Ni = v3load(cN,i), CAi = v3load(cCA,i), Ci = v3load(cC,i);
  float phi = 0.f, psi = 0.f, om = 0.f;
  if (i > 0)      phi = dihedral(v3load(cC,i-1), Ni, CAi, Ci);
  if (i < L_-1) { psi = dihedral(Ni, CAi, Ci, v3load(cN,i+1));
                  om  = dihedral(CAi, Ci, v3load(cN,i+1), v3load(cCA,i+1)); }
  float* o = nf + (size_t)i*15;
  o[0]=sinf(phi); o[1]=cosf(phi); o[2]=sinf(psi); o[3]=cosf(psi); o[4]=sinf(om); o[5]=cosf(om);
  V3 x = vunit(vsub(Ci, CAi));
  V3 v = vsub(Ni, CAi);
  V3 z = vunit(vcross(x, v));
  V3 y = vcross(z, x);
  // R = stack([x,y,z], axis=-1), reshape(9): element 3r+c = basis_c[r]
  o[6]=x.x;  o[7]=y.x;  o[8]=z.x;
  o[9]=x.y;  o[10]=y.y; o[11]=z.y;
  o[12]=x.z; o[13]=y.z; o[14]=z.z;
}

// ======================= edge features [E,324] (f16) =======================
__global__ void edge_feats_kernel(const float* __restrict__ cN, const float* __restrict__ cCA,
                                  const float* __restrict__ cC, const float* __restrict__ cO,
                                  const int* __restrict__ dstIdx, const float* __restrict__ nf,
                                  half_t* __restrict__ ef) {
  int e = blockIdx.x*blockDim.x + threadIdx.x;
  if (e >= E_) return;
  int s = e / KNN_;
  int d = dstIdx[e];
  V3 bs[4] = { v3load(cN,s), v3load(cCA,s), v3load(cC,s), v3load(cO,s) };
  V3 bd[4] = { v3load(cN,d), v3load(cCA,d), v3load(cC,d), v3load(cO,d) };
  half_t* o = ef + (size_t)e * EDGE_;
  int c = 0;
  for (int ai = 0; ai < 4; ++ai)
    for (int aj = 0; aj < 4; ++aj) {
      V3 df = vsub(bs[ai], bd[aj]);
      float dist = sqrtf(vdot(df, df));
      for (int r = 0; r < 16; ++r) {
        float cen = 20.0f * (float)r / 15.0f;
        float t = dist - cen;
        o[c++] = (half_t)expf(-0.64f * t * t);
      }
    }
  int sep = d - s;
  sep = sep < -32 ? -32 : (sep > 32 ? 32 : sep);
  sep += 32;
  for (int k = 0; k < 65; ++k) o[256 + k] = (half_t)((k == sep) ? 1.0f : 0.0f);
  V3 dir = vsub(bd[1], bs[1]);   // CA[dst]-CA[src]
  const float* R = nf + (size_t)s*15 + 6;   // R[r][c] = R[3r+c]
  float v0 = R[0]*dir.x + R[3]*dir.y + R[6]*dir.z;
  float v1 = R[1]*dir.x + R[4]*dir.y + R[7]*dir.z;
  float v2 = R[2]*dir.x + R[5]*dir.y + R[8]*dir.z;
  float n = sqrtf(v0*v0 + v1*v1 + v2*v2) + 1e-8f;
  o[321] = (half_t)(v0/n); o[322] = (half_t)(v1/n); o[323] = (half_t)(v2/n);
}

// ======================= WMMA GEMM: D = act(A[M,K] * W[N,K]^T + b) =======================
// A: f16 row-major, lda = K padded to mult of 32.  W: f16 row-major [N, K].
// Register-blocked: each wave computes an MT*16 x NT*16 output block with
// MT*NT chained v_wmma_f32_16x16x32_f16 per 32-wide K chunk (A frags reused
// across NT column tiles, B frags across MT row tiles).
template<int MT, int NT, bool RELU, bool OUT_F16>
__global__ void __launch_bounds__(128)
wmma_gemm_kernel(const half_t* __restrict__ A, int lda,
                 const half_t* __restrict__ B, int ldb,
                 const float* __restrict__ bias,
                 void* __restrict__ C, int ldc, int M, int N)
{
  int wave = threadIdx.x >> 5, lane = threadIdx.x & 31;
  int m0 = (blockIdx.x*4 + wave) * (16*MT);
  int n0 = blockIdx.y * (16*NT);
  if (m0 >= M) return;
  int hs = lane >> 4;            // lane half-select
  const half_t* arow[MT];
  #pragma unroll
  for (int i = 0; i < MT; ++i)
    arow[i] = A + (size_t)(m0 + i*16 + (lane & 15))*lda + hs*16;
  const half_t* brow[NT];
  #pragma unroll
  for (int j = 0; j < NT; ++j)
    brow[j] = B + (size_t)(n0 + j*16 + (lane & 15))*ldb + hs*16;
  v8f zero = {};
  v8f acc[MT][NT];
  #pragma unroll
  for (int i = 0; i < MT; ++i)
    #pragma unroll
    for (int j = 0; j < NT; ++j) acc[i][j] = zero;
  for (int k = 0; k < lda; k += 32) {
    v16h a[MT], b[NT];
    #pragma unroll
    for (int i = 0; i < MT; ++i) {
      // A frag: lanes 0-15 hold K k..k+15 ; lanes 16-31 hold k+16..k+31
      a[i] = *(const v16h*)(arow[i] + k);
    }
    #pragma unroll
    for (int j = 0; j < NT; ++j)
      b[j] = *(const v16h*)(brow[j] + k);   // lane holds column n, 16 contiguous K
    #pragma unroll
    for (int i = 0; i < MT; ++i)
      #pragma unroll
      for (int j = 0; j < NT; ++j)
        acc[i][j] = __builtin_amdgcn_wmma_f32_16x16x32_f16(false, a[i], false, b[j],
                                                           (short)0, acc[i][j], false, false);
  }
  #pragma unroll
  for (int j = 0; j < NT; ++j) {
    int col = n0 + j*16 + (lane & 15);
    float bv = bias ? bias[col] : 0.0f;
    #pragma unroll
    for (int i = 0; i < MT; ++i) {
      int mrow = m0 + i*16 + hs*8;   // D layout: vgpr r -> row r (lanes<16) / 8+r (lanes>=16)
      #pragma unroll
      for (int r = 0; r < 8; ++r) {
        float v = acc[i][j][r] + bv;
        if (RELU) v = fmaxf(v, 0.0f);
        if (OUT_F16) ((half_t*)C)[(size_t)(mrow + r)*ldc + col] = (half_t)v;
        else         ((float*)C)[(size_t)(mrow + r)*ldc + col] = v;
      }
    }
  }
}

// ======================= small utility kernels =======================
__global__ void cast_pad_kernel(const float* __restrict__ src, int src_ld, int ncols,
                                half_t* __restrict__ dst, int dst_ld, int wcols, int total) {
  int i = blockIdx.x*256 + threadIdx.x;
  if (i >= total) return;
  int r = i / wcols, c = i % wcols;
  dst[(size_t)r*dst_ld + c] = (c < ncols) ? (half_t)src[(size_t)r*src_ld + c] : (half_t)0.0f;
}

__global__ void transpose_cast_kernel(const float* __restrict__ src, int src_ld, int cols,
                                      half_t* __restrict__ dst, int dst_ld, int total) {
  int i = blockIdx.x*256 + threadIdx.x;
  if (i >= total) return;
  int r = i / cols, c = i % cols;
  dst[(size_t)c*dst_ld + r] = (half_t)src[(size_t)r*src_ld + c];
}

__global__ void build_msgin_kernel(const float* __restrict__ h, const half_t* __restrict__ ef,
                                   const int* __restrict__ dstIdx, half_t* __restrict__ out, int total) {
  int i = blockIdx.x*256 + threadIdx.x;
  if (i >= total) return;
  int e = i / MSGK_, c = i % MSGK_;
  half_t v;
  if (c < HID_)              v = (half_t)h[(size_t)(e / KNN_)*HID_ + c];
  else if (c < 2*HID_)       v = (half_t)h[(size_t)dstIdx[e]*HID_ + (c - HID_)];
  else if (c < 2*HID_+EDGE_) v = ef[(size_t)e*EDGE_ + (c - 2*HID_)];
  else                       v = (half_t)0.0f;
  out[i] = v;
}

__global__ void fill_zero_kernel(float* p, int n) {
  int i = blockIdx.x*256 + threadIdx.x;
  if (i < n) p[i] = 0.0f;
}

__global__ void segsum_kernel(const float* __restrict__ m, const int* __restrict__ dstIdx,
                              float* __restrict__ agg, int total) {
  int i = blockIdx.x*256 + threadIdx.x;
  if (i >= total) return;
  int e = i / HID_, c = i % HID_;
  atomicAdd(&agg[(size_t)dstIdx[e]*HID_ + c], m[i]);
}

__global__ void add_ln_kernel(float* __restrict__ h, const float* __restrict__ u,
                              const float* __restrict__ g, const float* __restrict__ b) {
  __shared__ float red[256];
  __shared__ float s_mean, s_inv;
  int r = blockIdx.x, t = threadIdx.x;
  float v = 0.0f;
  if (t < HID_) v = h[(size_t)r*HID_ + t] + u[(size_t)r*HID_ + t];
  red[t] = (t < HID_) ? v : 0.0f;
  __syncthreads();
  for (int s = 128; s > 0; s >>= 1) { if (t < s) red[t] += red[t+s]; __syncthreads(); }
  if (t == 0) s_mean = red[0] / (float)HID_;
  __syncthreads();
  float d = (t < HID_) ? (v - s_mean) : 0.0f;
  red[t] = d*d;
  __syncthreads();
  for (int s = 128; s > 0; s >>= 1) { if (t < s) red[t] += red[t+s]; __syncthreads(); }
  if (t == 0) s_inv = rsqrtf(red[0] / (float)HID_ + 1e-5f);
  __syncthreads();
  if (t < HID_) h[(size_t)r*HID_ + t] = (v - s_mean)*s_inv*g[t] + b[t];
}

__global__ void rank_kernel(const int* __restrict__ order, int* __restrict__ rank) {
  int i = blockIdx.x*256 + threadIdx.x;
  if (i < L_) rank[order[i]] = i;
}

__global__ void tok_kernel(const int* __restrict__ tokens, const float* __restrict__ emb,
                           float* __restrict__ hdec, int total) {
  int i = blockIdx.x*256 + threadIdx.x;
  if (i >= total) return;
  int r = i / HID_, c = i % HID_;
  hdec[i] = emb[(size_t)tokens[r]*HID_ + c];
}

__global__ void softmax_kernel(const float* __restrict__ S, const int* __restrict__ rankv,
                               const unsigned char* __restrict__ mask, int causal, float scale,
                               half_t* __restrict__ P) {
  __shared__ float buf[L_];
  __shared__ float red[256];
  int q = blockIdx.x, t = threadIdx.x;
  const float* srow = S + (size_t)q * L_;
  int rq = rankv[q];
  float mx = -3.4e38f;
  for (int k = t; k < L_; k += 256) {
    float l = srow[k] * scale;
    if (causal && !((rankv[k] <= rq) && (mask[k] != 0))) l += -1.0e9f;
    buf[k] = l;
    mx = fmaxf(mx, l);
  }
  red[t] = mx;
  __syncthreads();
  for (int s = 128; s > 0; s >>= 1) { if (t < s) red[t] = fmaxf(red[t], red[t+s]); __syncthreads(); }
  float m = red[0];
  __syncthreads();
  float sum = 0.0f;
  for (int k = t; k < L_; k += 256) { float ev = __expf(buf[k] - m); buf[k] = ev; sum += ev; }
  red[t] = sum;
  __syncthreads();
  for (int s = 128; s > 0; s >>= 1) { if (t < s) red[t] += red[t+s]; __syncthreads(); }
  float inv = 1.0f / red[0];
  for (int k = t; k < L_; k += 256) P[(size_t)q*L_ + k] = (half_t)(buf[k] * inv);
}

__global__ void final_proj_kernel(const float* __restrict__ h, const float* __restrict__ w,
                                  const float* __restrict__ b, float* __restrict__ out, int total) {
  int i = blockIdx.x*256 + threadIdx.x;
  if (i >= total) return;
  int r = i / 20, v = i % 20;
  float acc = b[v];
  const float* hr = h + (size_t)r*HID_;
  const float* wr = w + (size_t)v*HID_;
  for (int c = 0; c < HID_; ++c) acc += hr[c] * wr[c];
  out[i] = acc;
}

// ======================= host orchestration =======================
// d_in index map (jax tree-flatten of setup_inputs dict; params dict keys sorted):
//  0..3 coords N/CA/C/O, 4 tokens(i32), 5 decoding_order(i32), 6 mask(bool8),
//  7..60   dec: 3 layers x 18 (ca_o_b, ca_o_w, ca_qkv_b, ca_qkv_w, f1b, f1w, f2b, f2w,
//                              n1b, n1g, n2b, n2g, n3b, n3g, sa_o_b, sa_o_w, sa_qkv_b, sa_qkv_w)
//  61..90  enc: 3 layers x 10 (m1b, m1w, m2b, m2w, nb, ng, u1b, u1w, u2b, u2w)
//  91 node_emb_b, 92 node_emb_w, 93 out_b, 94 out_w, 95 tok_emb
enum { D_CA_O_B=0, D_CA_O_W, D_CA_QKV_B, D_CA_QKV_W, D_F1B, D_F1W, D_F2B, D_F2W,
       D_N1B, D_N1G, D_N2B, D_N2G, D_N3B, D_N3G, D_SA_O_B, D_SA_O_W, D_SA_QKV_B, D_SA_QKV_W };
enum { P_M1B=0, P_M1W, P_M2B, P_M2W, P_NB, P_NG, P_U1B, P_U1W, P_U2B, P_U2W };

extern "C" void kernel_launch(void* const* d_in, const int* in_sizes, int n_in,
                              void* d_out, int out_size, void* d_ws, size_t ws_size,
                              hipStream_t stream) {
  (void)in_sizes; (void)n_in; (void)out_size; (void)ws_size;
  const int DEC0 = 7, ENC0 = 61, NODE_B = 91, NODE_W = 92, OUT_B = 93, OUT_W = 94, TOK_EMB = 95;
  auto F    = [&](int i)         { return (const float*)d_in[i]; };
  auto decp = [&](int l, int f)  { return (const float*)d_in[DEC0 + l*18 + f]; };
  auto encp = [&](int l, int f)  { return (const float*)d_in[ENC0 + l*10 + f]; };
  const float* cN  = F(0); const float* cCA = F(1); const float* cC = F(2); const float* cO = F(3);
  const int* tokens = (const int*)d_in[4];
  const int* dorder = (const int*)d_in[5];
  const unsigned char* maskp = (const unsigned char*)d_in[6];

  // ---------------- workspace bump allocator ----------------
  char* wp = (char*)d_ws;
  auto alloc = [&](size_t n) { char* r = wp; wp += (n + 255) & ~(size_t)255; return r; };
  int*    dstIdx = (int*)   alloc((size_t)E_*4);
  int*    rank   = (int*)   alloc((size_t)L_*4);
  float*  nf     = (float*) alloc((size_t)L_*15*4);
  half_t* nf16   = (half_t*)alloc((size_t)L_*32*2);
  half_t* ef16   = (half_t*)alloc((size_t)E_*EDGE_*2);
  float*  henc   = (float*) alloc((size_t)L_*HID_*4);
  float*  hdec   = (float*) alloc((size_t)L_*HID_*4);
  half_t* henc16 = (half_t*)alloc((size_t)L_*HID_*2);
  half_t* hdec16 = (half_t*)alloc((size_t)L_*HID_*2);
  half_t* msgin16= (half_t*)alloc((size_t)E_*MSGK_*2);
  half_t* m1o16  = (half_t*)alloc((size_t)E_*HID_*2);
  float*  msum   = (float*) alloc((size_t)E_*HID_*4);
  float*  agg    = (float*) alloc((size_t)L_*HID_*4);
  half_t* cat216 = (half_t*)alloc((size_t)L_*2*HID_*2);
  half_t* act16  = (half_t*)alloc((size_t)L_*4*HID_*2);   // relu intermediates (192 or 768 wide)
  float*  fbuf   = (float*) alloc((size_t)L_*HID_*4);     // u / proj f32
  float*  qkvbuf = (float*) alloc((size_t)L_*3*HID_*4);
  float*  qbuf   = (float*) alloc((size_t)L_*HID_*4);
  float*  kvbuf  = (float*) alloc((size_t)L_*2*HID_*4);
  half_t* Qh16   = (half_t*)alloc((size_t)L_*DHP_*2);
  half_t* Kh16   = (half_t*)alloc((size_t)L_*DHP_*2);
  half_t* Vt16   = (half_t*)alloc((size_t)DH_*L_*2);
  float*  Sbuf   = (float*) alloc((size_t)L_*L_*4);
  half_t* P16    = (half_t*)alloc((size_t)L_*L_*2);
  float*  attnO  = (float*) alloc((size_t)L_*HID_*4);
  half_t* attnO16= (half_t*)alloc((size_t)L_*HID_*2);
  half_t* wnode16= (half_t*)alloc((size_t)HID_*32*2);
  half_t *encm1[3], *encm2[3], *encu1[3], *encu2[3];
  half_t *saqkv[3], *sao[3], *caqkv[3], *cao[3], *f1w[3], *f2w[3];
  for (int l = 0; l < 3; ++l) {
    encm1[l] = (half_t*)alloc((size_t)HID_*MSGK_*2);
    encm2[l] = (half_t*)alloc((size_t)HID_*HID_*2);
    encu1[l] = (half_t*)alloc((size_t)HID_*2*HID_*2);
    encu2[l] = (half_t*)alloc((size_t)HID_*HID_*2);
    saqkv[l] = (half_t*)alloc((size_t)3*HID_*HID_*2);
    sao[l]   = (half_t*)alloc((size_t)HID_*HID_*2);
    caqkv[l] = (half_t*)alloc((size_t)3*HID_*HID_*2);
    cao[l]   = (half_t*)alloc((size_t)HID_*HID_*2);
    f1w[l]   = (half_t*)alloc((size_t)4*HID_*HID_*2);
    f2w[l]   = (half_t*)alloc((size_t)HID_*4*HID_*2);
  }

  auto castp = [&](const float* src, int sld, int nc, half_t* dst, int dld, int wc, int M) {
    int total = M * wc;
    cast_pad_kernel<<<(total + 255)/256, 256, 0, stream>>>(src, sld, nc, dst, dld, wc, total);
  };
  // Register-blocked WMMA GEMM dispatch: 32x64 per wave (2x3 for N=48).
  auto gemm = [&](const half_t* A, int lda, const half_t* B, int ldb, const float* bias,
                  void* C, int ldc, int M, int N, bool relu_f16) {
    if (N % 64 == 0) {
      dim3 grid((M + 127)/128, N/64);
      if (relu_f16) wmma_gemm_kernel<2,4,true, true ><<<grid,128,0,stream>>>(A,lda,B,ldb,bias,C,ldc,M,N);
      else          wmma_gemm_kernel<2,4,false,false><<<grid,128,0,stream>>>(A,lda,B,ldb,bias,C,ldc,M,N);
    } else {  // N == 48
      dim3 grid((M + 127)/128, N/48);
      wmma_gemm_kernel<2,3,false,false><<<grid,128,0,stream>>>(A,lda,B,ldb,bias,C,ldc,M,N);
    }
  };

  // ---------------- weights f32 -> f16 (padded K) ----------------
  castp(F(NODE_W), 15, 15, wnode16, 32, 32, HID_);
  for (int l = 0; l < 3; ++l) {
    castp(encp(l,P_M1W), 708, 708, encm1[l], MSGK_, MSGK_, HID_);
    castp(encp(l,P_M2W), HID_, HID_, encm2[l], HID_, HID_, HID_);
    castp(encp(l,P_U1W), 2*HID_, 2*HID_, encu1[l], 2*HID_, 2*HID_, HID_);
    castp(encp(l,P_U2W), HID_, HID_, encu2[l], HID_, HID_, HID_);
    castp(decp(l,D_SA_QKV_W), HID_, HID_, saqkv[l], HID_, HID_, 3*HID_);
    castp(decp(l,D_SA_O_W),   HID_, HID_, sao[l],   HID_, HID_, HID_);
    castp(decp(l,D_CA_QKV_W), HID_, HID_, caqkv[l], HID_, HID_, 3*HID_);
    castp(decp(l,D_CA_O_W),   HID_, HID_, cao[l],   HID_, HID_, HID_);
    castp(decp(l,D_F1W), HID_, HID_, f1w[l], HID_, HID_, 4*HID_);
    castp(decp(l,D_F2W), 4*HID_, 4*HID_, f2w[l], 4*HID_, 4*HID_, HID_);
  }

  // ---------------- geometry ----------------
  knn_kernel<<<L_, 256, 0, stream>>>(cCA, dstIdx);
  node_feats_kernel<<<(L_ + 255)/256, 256, 0, stream>>>(cN, cCA, cC, nf);
  edge_feats_kernel<<<(E_ + 255)/256, 256, 0, stream>>>(cN, cCA, cC, cO, dstIdx, nf, ef16);

  // ---------------- node embedding ----------------
  castp(nf, 15, 15, nf16, 32, 32, L_);
  gemm(nf16, 32, wnode16, 32, F(NODE_B), henc, HID_, L_, HID_, false);

  // ---------------- encoder (MPNN) ----------------
  for (int l = 0; l < 3; ++l) {
    build_msgin_kernel<<<(E_*MSGK_ + 255)/256, 256, 0, stream>>>(henc, ef16, dstIdx, msgin16, E_*MSGK_);
    gemm(msgin16, MSGK_, encm1[l], MSGK_, encp(l,P_M1B), m1o16, HID_, E_, HID_, true);
    gemm(m1o16, HID_, encm2[l], HID_, encp(l,P_M2B), msum, HID_, E_, HID_, false);
    fill_zero_kernel<<<(L_*HID_ + 255)/256, 256, 0, stream>>>(agg, L_*HID_);
    segsum_kernel<<<(E_*HID_ + 255)/256, 256, 0, stream>>>(msum, dstIdx, agg, E_*HID_);
    castp(henc, HID_, HID_, cat216,        2*HID_, HID_, L_);
    castp(agg,  HID_, HID_, cat216 + HID_, 2*HID_, HID_, L_);
    gemm(cat216, 2*HID_, encu1[l], 2*HID_, encp(l,P_U1B), act16, HID_, L_, HID_, true);
    gemm(act16, HID_, encu2[l], HID_, encp(l,P_U2B), fbuf, HID_, L_, HID_, false);
    add_ln_kernel<<<L_, 256, 0, stream>>>(henc, fbuf, encp(l,P_NG), encp(l,P_NB));
  }

  // ---------------- decoder setup ----------------
  rank_kernel<<<(L_ + 255)/256, 256, 0, stream>>>(dorder, rank);
  tok_kernel<<<(L_*HID_ + 255)/256, 256, 0, stream>>>(tokens, F(TOK_EMB), hdec, L_*HID_);
  castp(henc, HID_, HID_, henc16, HID_, HID_, L_);
  castp(hdec, HID_, HID_, hdec16, HID_, HID_, L_);

  const float scale = 0.14433756729740643f;  // 1/sqrt(48)
  auto attention = [&](const float* qsrc, int qld, const float* ksrc, int kld,
                       const float* vsrc, int vld, int causal) {
    for (int hh = 0; hh < 4; ++hh) {
      castp(qsrc + hh*DH_, qld, DH_, Qh16, DHP_, DHP_, L_);
      castp(ksrc + hh*DH_, kld, DH_, Kh16, DHP_, DHP_, L_);
      gemm(Qh16, DHP_, Kh16, DHP_, nullptr, Sbuf, L_, L_, L_, false);
      softmax_kernel<<<L_, 256, 0, stream>>>(Sbuf, rank, maskp, causal, scale, P16);
      transpose_cast_kernel<<<(L_*DH_ + 255)/256, 256, 0, stream>>>(vsrc + hh*DH_, vld, DH_, Vt16, L_, L_*DH_);
      gemm(P16, L_, Vt16, L_, nullptr, attnO + hh*DH_, HID_, L_, DH_, false);
    }
  };

  // ---------------- decoder layers ----------------
  for (int l = 0; l < 3; ++l) {
    // self-attention (causal by decoding order)
    gemm(hdec16, HID_, saqkv[l], HID_, decp(l,D_SA_QKV_B), qkvbuf, 3*HID_, L_, 3*HID_, false);
    attention(qkvbuf, 3*HID_, qkvbuf + HID_, 3*HID_, qkvbuf + 2*HID_, 3*HID_, 1);
    castp(attnO, HID_, HID_, attnO16, HID_, HID_, L_);
    gemm(attnO16, HID_, sao[l], HID_, decp(l,D_SA_O_B), fbuf, HID_, L_, HID_, false);
    add_ln_kernel<<<L_, 256, 0, stream>>>(hdec, fbuf, decp(l,D_N1G), decp(l,D_N1B));
    castp(hdec, HID_, HID_, hdec16, HID_, HID_, L_);
    // cross-attention (q from hdec, k/v from encoder h)
    gemm(hdec16, HID_, caqkv[l], HID_, decp(l,D_CA_QKV_B), qbuf, HID_, L_, HID_, false);
    gemm(henc16, HID_, caqkv[l] + (size_t)HID_*HID_, HID_, decp(l,D_CA_QKV_B) + HID_,
         kvbuf, 2*HID_, L_, 2*HID_, false);
    attention(qbuf, HID_, kvbuf, 2*HID_, kvbuf + HID_, 2*HID_, 0);
    castp(attnO, HID_, HID_, attnO16, HID_, HID_, L_);
    gemm(attnO16, HID_, cao[l], HID_, decp(l,D_CA_O_B), fbuf, HID_, L_, HID_, false);
    add_ln_kernel<<<L_, 256, 0, stream>>>(hdec, fbuf, decp(l,D_N2G), decp(l,D_N2B));
    castp(hdec, HID_, HID_, hdec16, HID_, HID_, L_);
    // FFN
    gemm(hdec16, HID_, f1w[l], HID_, decp(l,D_F1B), act16, 4*HID_, L_, 4*HID_, true);
    gemm(act16, 4*HID_, f2w[l], 4*HID_, decp(l,D_F2B), fbuf, HID_, L_, HID_, false);
    add_ln_kernel<<<L_, 256, 0, stream>>>(hdec, fbuf, decp(l,D_N3G), decp(l,D_N3B));
    castp(hdec, HID_, HID_, hdec16, HID_, HID_, L_);
  }

  // ---------------- output projection [L,20] ----------------
  final_proj_kernel<<<(L_*20 + 255)/256, 256, 0, stream>>>(hdec, F(OUT_W), F(OUT_B), (float*)d_out, L_*20);
}
